// Longformer_13615046328322
// MI455X (gfx1250) — compile-verified, matrix operand order checked by reading, not run on hardware
//
#include <hip/hip_runtime.h>
#include <hip/hip_bf16.h>

#define BB 2
#define SS 4096
#define HH 12
#define DHD 64
#define DD 768
#define WW 256
#define GG 64
#define NCHUNK 16

typedef _Float16 v16h __attribute__((ext_vector_type(16)));
typedef _Float16 v4h  __attribute__((ext_vector_type(4)));
typedef float    v8f  __attribute__((ext_vector_type(8)));
typedef unsigned int v4u __attribute__((ext_vector_type(4)));
typedef int      v8i  __attribute__((ext_vector_type(8)));
typedef int      v4i  __attribute__((ext_vector_type(4)));

// ---------------------------------------------------------------------------
// WMMA fragment helpers (CDNA5 wave32 layouts, CDNA5 ISA)
// A 16x32 f16: lane&15 = row M; lanes<16 hold K {0..7,16..23}, lanes>=16 hold
// K {8..15,24..31}.  B 32x16 f16: lane&15 = column N, same K split.
// C/D 16x16 f32: lane&15 = col N, VGPR r = row r (+8 for lanes>=16).
// ---------------------------------------------------------------------------
__device__ __forceinline__ v16h frag_rowptr(const _Float16* rowp, int lane) {
  const int lo = (lane >> 4) << 3;   // 0 or 8
  v16h a;
#pragma unroll
  for (int j = 0; j < 8; ++j) { a[j] = rowp[lo + j]; a[8 + j] = rowp[16 + lo + j]; }
  return a;
}

__device__ __forceinline__ v16h frag_tile(const _Float16* tilebase, int stride, int lane) {
  return frag_rowptr(tilebase + (size_t)(lane & 15) * stride, lane);
}

__device__ __forceinline__ v8f wmma_f16(v16h a, v16h b, v8f c) {
  return __builtin_amdgcn_wmma_f32_16x16x32_f16(false, a, false, b, (short)0, c,
                                                false, false);
}

__device__ __forceinline__ v8f zero8() {
  v8f z;
#pragma unroll
  for (int r = 0; r < 8; ++r) z[r] = 0.f;
  return z;
}

__device__ __forceinline__ v16h ones16h() {
  v16h o;
#pragma unroll
  for (int j = 0; j < 16; ++j) o[j] = (_Float16)1.0f;
  return o;
}

// XOR butterfly max within 16-lane halves: ds_swizzle (group-of-32 XOR mode,
// and_mask=0x1F) when available, else __shfl_xor (lowers to ds_bpermute).
template <int OFF>
__device__ __forceinline__ float bflymax(float v) {
#if __has_builtin(__builtin_amdgcn_ds_swizzle)
  const int o = __builtin_amdgcn_ds_swizzle(__float_as_int(v), 0x7C00 | OFF);
  return fmaxf(v, __int_as_float(o));
#else
  return fmaxf(v, __shfl_xor(v, OFF, 32));
#endif
}

// ---------------------------------------------------------------------------
// TDM probe: NULL descriptor (group0.count==0 => NULL tensor, NOP per D# spec)
// exercises tensor_load_to_lds + s_wait_tensorcnt + cluster barrier safely.
// ---------------------------------------------------------------------------
__global__ void tdm_probe_kernel() {
  v4u g0 = {0u, 0u, 0u, 0u};
  v8i g1 = {0, 0, 0, 0, 0, 0, 0, 0};
  v4i g2 = {0, 0, 0, 0};
  v4i g3 = {0, 0, 0, 0};
#if defined(__clang_major__) && (__clang_major__ >= 23)
  v8i g4 = {0, 0, 0, 0, 0, 0, 0, 0};
  __builtin_amdgcn_tensor_load_to_lds(g0, g1, g2, g3, g4, 0);
#else
  __builtin_amdgcn_tensor_load_to_lds(g0, g1, g2, g3, 0);
#endif
  __builtin_amdgcn_s_wait_tensorcnt(0);
  __builtin_amdgcn_s_cluster_barrier();   // NOP when not in a cluster
}

// ---------------------------------------------------------------------------
// h = emb[x] + pos  (f32 -> f16), vectorized
// ---------------------------------------------------------------------------
__global__ __launch_bounds__(192) void embed_kernel(
    const int* __restrict__ x, const float* __restrict__ emb,
    const float* __restrict__ pos, _Float16* __restrict__ hf) {
  const int row = blockIdx.x;            // 0 .. B*S-1
  const int s = row % SS;
  const int tok = x[row];
  const float4* e = (const float4*)(emb + (size_t)tok * DD);
  const float4* p = (const float4*)(pos + (size_t)s * DD);
  v4h* o = (v4h*)(hf + (size_t)row * DD);
  const int j = threadIdx.x;             // 0..191 covers 768 floats
  const float4 ev = e[j], pv = p[j];
  v4h r;
  r[0] = (_Float16)(ev.x + pv.x);
  r[1] = (_Float16)(ev.y + pv.y);
  r[2] = (_Float16)(ev.z + pv.z);
  r[3] = (_Float16)(ev.w + pv.w);
  o[j] = r;
}

// ---------------------------------------------------------------------------
// weight transpose f32[K][N] -> f16 wT[N][K]
// ---------------------------------------------------------------------------
__global__ __launch_bounds__(256) void transpose_kernel(
    const float* __restrict__ w, _Float16* __restrict__ wt) {
  __shared__ float tile[16][17];
  const int tx = threadIdx.x & 15, ty = threadIdx.x >> 4;
  const int k = blockIdx.y * 16 + ty;
  const int n = blockIdx.x * 16 + tx;
  tile[ty][tx] = w[(size_t)k * DD + n];
  __syncthreads();
  const int on = blockIdx.x * 16 + ty;
  const int ok = blockIdx.y * 16 + tx;
  wt[(size_t)on * DD + ok] = (_Float16)tile[tx][ty];
}

__global__ void flags_zero_kernel(int* f) {
  int i = blockIdx.x * 256 + threadIdx.x;
  if (i < SS) f[i] = 0;
}
__global__ void flags_set_kernel(const int* __restrict__ clss, int* f) {
  f[clss[threadIdx.x]] = 1;
}

// ---------------------------------------------------------------------------
// GEMM: C[M x 768] = alpha * A[M x 768] @ B[768 x 768], B given transposed
// (Bt[n][k]).  128x128 block tile, BK=32, 8 waves of 32x64, double-buffered.
// ---------------------------------------------------------------------------
template <bool OUT_F32>
__global__ __launch_bounds__(256) void gemm768_kernel(
    const _Float16* __restrict__ A, const _Float16* __restrict__ Bt,
    void* __restrict__ Cout, float alpha) {
  __shared__ _Float16 As[2][128 * 32];
  __shared__ _Float16 Bs[2][128 * 32];
  const int tid = threadIdx.x, lane = tid & 31, wid = tid >> 5;
  const int mw = wid & 3, nw = wid >> 2;
  const int m0 = blockIdx.x * 128, n0 = blockIdx.y * 128;
  const int hi = lane >> 4, lc = lane & 15;

  v8f acc[2][4];
#pragma unroll
  for (int i = 0; i < 2; ++i)
#pragma unroll
    for (int j = 0; j < 4; ++j) acc[i][j] = zero8();

  const int lr = tid >> 1;             // 0..127: staged row
  const int lcc = (tid & 1) * 16;      // 0 or 16 (halfs)
  const _Float16* ga = A + (size_t)(m0 + lr) * DD + lcc;
  const _Float16* gb = Bt + (size_t)(n0 + lr) * DD + lcc;

  uint4 ra0 = *(const uint4*)(ga);
  uint4 ra1 = *(const uint4*)(ga + 8);
  uint4 rb0 = *(const uint4*)(gb);
  uint4 rb1 = *(const uint4*)(gb + 8);
  *(uint4*)(&As[0][lr * 32 + lcc]) = ra0;
  *(uint4*)(&As[0][lr * 32 + lcc + 8]) = ra1;
  *(uint4*)(&Bs[0][lr * 32 + lcc]) = rb0;
  *(uint4*)(&Bs[0][lr * 32 + lcc + 8]) = rb1;
  __syncthreads();

  for (int kt = 0; kt < 24; ++kt) {
    const int cur = kt & 1;
    if (kt + 1 < 24) {
      const int k1 = (kt + 1) * 32;
      ra0 = *(const uint4*)(ga + k1);
      ra1 = *(const uint4*)(ga + k1 + 8);
      rb0 = *(const uint4*)(gb + k1);
      rb1 = *(const uint4*)(gb + k1 + 8);
      if (kt + 2 < 24) {
        __builtin_prefetch(ga + (kt + 2) * 32, 0, 1);   // global_prefetch_b8
        __builtin_prefetch(gb + (kt + 2) * 32, 0, 1);
      }
    }
    v16h af[2], bf[4];
#pragma unroll
    for (int mi = 0; mi < 2; ++mi)
      af[mi] = frag_tile(&As[cur][(mw * 32 + mi * 16) * 32], 32, lane);
#pragma unroll
    for (int ni = 0; ni < 4; ++ni)
      bf[ni] = frag_tile(&Bs[cur][(nw * 64 + ni * 16) * 32], 32, lane);
#pragma unroll
    for (int mi = 0; mi < 2; ++mi)
#pragma unroll
      for (int ni = 0; ni < 4; ++ni)
        acc[mi][ni] = wmma_f16(af[mi], bf[ni], acc[mi][ni]);
    __syncthreads();
    if (kt + 1 < 24) {
      const int nxt = cur ^ 1;
      *(uint4*)(&As[nxt][lr * 32 + lcc]) = ra0;
      *(uint4*)(&As[nxt][lr * 32 + lcc + 8]) = ra1;
      *(uint4*)(&Bs[nxt][lr * 32 + lcc]) = rb0;
      *(uint4*)(&Bs[nxt][lr * 32 + lcc + 8]) = rb1;
      __syncthreads();
    }
  }

#pragma unroll
  for (int mi = 0; mi < 2; ++mi)
#pragma unroll
    for (int ni = 0; ni < 4; ++ni)
#pragma unroll
      for (int r = 0; r < 8; ++r) {
        const int row = m0 + mw * 32 + mi * 16 + r + 8 * hi;
        const int col = n0 + nw * 64 + ni * 16 + lc;
        const float v = acc[mi][ni][r] * alpha;
        if (OUT_F32)
          ((float*)Cout)[(size_t)row * DD + col] = v;
        else
          ((_Float16*)Cout)[(size_t)row * DD + col] = (_Float16)v;
      }
}

// ---------------------------------------------------------------------------
// Flash-softmax tile update shared by both attention kernels.
// Row sums come from an extra WMMA against an all-ones B matrix (replicated
// across lanes in C layout), so only the max needs a cross-lane butterfly.
// ---------------------------------------------------------------------------
#define VT_STRIDE 40   // halfs; 20 words -> conflict-free 16-step scatter

// ---------------------------------------------------------------------------
// Band + global-key attention with streaming (flash) softmax.
// Grid: B*H*NC*4 blocks of 128 threads; wave handles 16 queries.
// kt 0..23: band keys (window 3W); kt 24..25: the G=64 global keys.
// ---------------------------------------------------------------------------
__global__ __launch_bounds__(128) void band_attn_kernel(
    const _Float16* __restrict__ qf, const _Float16* __restrict__ kf,
    const _Float16* __restrict__ vf, const int* __restrict__ mask,
    const int* __restrict__ clss, const int* __restrict__ flags,
    _Float16* __restrict__ attn) {
  __shared__ _Float16 Vt[64][VT_STRIDE];     // [dh][key]
  __shared__ _Float16 Pb[4][16][VT_STRIDE];  // per-wave P tile
  int t = blockIdx.x;
  const int qt = t & 3; t >>= 2;
  const int c = t % NCHUNK; t /= NCHUNK;
  const int hh = t % HH;
  const int b = t / HH;
  const int tid = threadIdx.x, lane = tid & 31, wid = tid >> 5;
  const int hi = lane >> 4, lc = lane & 15;
  const int q0 = c * WW + qt * 64 + wid * 16;
  const size_t baseBS = (size_t)b * SS;

  const _Float16* qrow = qf + (baseBS + q0 + lc) * DD + hh * DHD;
  const v16h aq0 = frag_rowptr(qrow, lane);
  const v16h aq1 = frag_rowptr(qrow + 32, lane);
  const v16h ones = ones16h();

  float m8[8], l8[8];
  v8f acc[4];
#pragma unroll
  for (int r = 0; r < 8; ++r) { m8[r] = -3.0e38f; l8[r] = 0.f; }
#pragma unroll
  for (int ni = 0; ni < 4; ++ni) acc[ni] = zero8();

  const int keyBase0 = c * WW - WW;
  const int vkey = tid >> 2;           // staging key 0..31
  const int vseg = tid & 3;            // staging dh segment

  for (int kt = 0; kt < 26; ++kt) {
    const bool gmode = (kt >= 24);
    __syncthreads();                    // prior-iter LDS reads done
    {
      int kpos;
      if (gmode) kpos = clss[(kt - 24) * 32 + vkey];
      else {
        kpos = keyBase0 + kt * 32 + vkey;
        kpos = kpos < 0 ? 0 : (kpos >= SS ? SS - 1 : kpos);
      }
      const uint4* vp = (const uint4*)(vf + (baseBS + kpos) * DD + hh * DHD +
                                       vseg * 16);
      union { uint4 u[2]; _Float16 h[16]; } cv;
      cv.u[0] = vp[0]; cv.u[1] = vp[1];
#pragma unroll
      for (int j = 0; j < 16; ++j) Vt[vseg * 16 + j][vkey] = cv.h[j];
    }
    __syncthreads();

    v8f s2[2];
#pragma unroll
    for (int u = 0; u < 2; ++u) {
      int kprel, krow;
      bool vk;
      if (gmode) {
        const int g = (kt - 24) * 32 + u * 16 + lc;
        krow = clss[g];
        kprel = krow;
        vk = (mask[baseBS + krow] != 0);
      } else {
        kprel = keyBase0 + kt * 32 + u * 16 + lc;
        krow = kprel < 0 ? 0 : (kprel >= SS ? SS - 1 : kprel);
        vk = ((unsigned)kprel < (unsigned)SS) && (mask[baseBS + krow] != 0) &&
             (flags[krow] == 0);
      }
      const _Float16* kr = kf + (baseBS + krow) * DD + hh * DHD;
      const v16h b0 = frag_rowptr(kr, lane);
      const v16h b1 = frag_rowptr(kr + 32, lane);
      v8f sc = zero8();
      sc = wmma_f16(aq0, b0, sc);
      sc = wmma_f16(aq1, b1, sc);
#pragma unroll
      for (int r = 0; r < 8; ++r) {
        const int qr = q0 + r + 8 * hi;
        const bool ok =
            vk && (gmode || ((unsigned)(kprel - qr + WW) <= (unsigned)(2 * WW)));
        sc[r] = ok ? sc[r] : -3.0e38f;
      }
      s2[u] = sc;
    }

    // running max (cross-lane butterfly within 16-lane halves)
    float rmx[8];
#pragma unroll
    for (int r = 0; r < 8; ++r) rmx[r] = fmaxf(s2[0][r], s2[1][r]);
#pragma unroll
    for (int r = 0; r < 8; ++r) rmx[r] = bflymax<1>(rmx[r]);
#pragma unroll
    for (int r = 0; r < 8; ++r) rmx[r] = bflymax<2>(rmx[r]);
#pragma unroll
    for (int r = 0; r < 8; ++r) rmx[r] = bflymax<4>(rmx[r]);
#pragma unroll
    for (int r = 0; r < 8; ++r) rmx[r] = bflymax<8>(rmx[r]);

    v8f p0, p1;
#pragma unroll
    for (int r = 0; r < 8; ++r) {
      const float nm = fmaxf(m8[r], rmx[r]);
      const float cf = __expf(m8[r] - nm);
      p0[r] = (s2[0][r] < -1.0e37f) ? 0.f : __expf(s2[0][r] - nm);
      p1[r] = (s2[1][r] < -1.0e37f) ? 0.f : __expf(s2[1][r] - nm);
      m8[r] = nm;
      l8[r] *= cf;
      acc[0][r] *= cf; acc[1][r] *= cf; acc[2][r] *= cf; acc[3][r] *= cf;
    }

    // bounce P (C layout) -> LDS -> A layout; acc += P @ V ; l += P @ ones
#pragma unroll
    for (int r = 0; r < 8; ++r) {
      Pb[wid][r + 8 * hi][lc] = (_Float16)p0[r];
      Pb[wid][r + 8 * hi][16 + lc] = (_Float16)p1[r];
    }
    __syncthreads();
    const v16h ap = frag_tile(&Pb[wid][0][0], VT_STRIDE, lane);
#pragma unroll
    for (int ni = 0; ni < 4; ++ni) {
      const v16h bv = frag_rowptr(&Vt[ni * 16 + lc][0], lane);
      acc[ni] = wmma_f16(ap, bv, acc[ni]);
    }
    const v8f lsum = wmma_f16(ap, ones, zero8());  // row sums, all lanes
#pragma unroll
    for (int r = 0; r < 8; ++r) l8[r] += lsum[r];
  }

#pragma unroll
  for (int ni = 0; ni < 4; ++ni)
#pragma unroll
    for (int r = 0; r < 8; ++r) {
      const int row = q0 + r + 8 * hi;
      const float invl = l8[r] > 0.f ? 1.0f / l8[r] : 0.f;
      attn[(baseBS + row) * DD + hh * DHD + ni * 16 + lc] =
          (_Float16)(acc[ni][r] * invl);
    }
}

// ---------------------------------------------------------------------------
// Global-query attention: G=64 clss queries attend over all S keys; the
// results overwrite the clss rows of `attn`.  Grid: B*H blocks, 128 threads.
// ---------------------------------------------------------------------------
__global__ __launch_bounds__(128) void global_attn_kernel(
    const _Float16* __restrict__ qgf, const _Float16* __restrict__ kgf,
    const _Float16* __restrict__ vgf, const int* __restrict__ mask,
    const int* __restrict__ clss, _Float16* __restrict__ attn) {
  __shared__ _Float16 Vt[64][VT_STRIDE];
  __shared__ _Float16 Pb[4][16][VT_STRIDE];
  const int hh = blockIdx.x % HH, b = blockIdx.x / HH;
  const int tid = threadIdx.x, lane = tid & 31, wid = tid >> 5;
  const int hi = lane >> 4, lc = lane & 15;
  const size_t baseBS = (size_t)b * SS;
  const int g0r = wid * 16;

  const int myq = clss[g0r + lc];
  const _Float16* qrow = qgf + (baseBS + myq) * DD + hh * DHD;
  const v16h aq0 = frag_rowptr(qrow, lane);
  const v16h aq1 = frag_rowptr(qrow + 32, lane);
  const v16h ones = ones16h();

  float m8[8], l8[8];
  v8f acc[4];
#pragma unroll
  for (int r = 0; r < 8; ++r) { m8[r] = -3.0e38f; l8[r] = 0.f; }
#pragma unroll
  for (int ni = 0; ni < 4; ++ni) acc[ni] = zero8();

  const int vkey = tid >> 2, vseg = tid & 3;

  for (int kt = 0; kt < SS / 32; ++kt) {
    __syncthreads();
    {
      const int kpos = kt * 32 + vkey;
      const uint4* vp = (const uint4*)(vgf + (baseBS + kpos) * DD + hh * DHD +
                                       vseg * 16);
      union { uint4 u[2]; _Float16 h[16]; } cv;
      cv.u[0] = vp[0]; cv.u[1] = vp[1];
#pragma unroll
      for (int j = 0; j < 16; ++j) Vt[vseg * 16 + j][vkey] = cv.h[j];
    }
    __syncthreads();

    v8f s2[2];
#pragma unroll
    for (int u = 0; u < 2; ++u) {
      const int kpos = kt * 32 + u * 16 + lc;
      const bool vk = (mask[baseBS + kpos] != 0);
      const _Float16* kr = kgf + (baseBS + kpos) * DD + hh * DHD;
      const v16h b0 = frag_rowptr(kr, lane);
      const v16h b1 = frag_rowptr(kr + 32, lane);
      v8f sc = zero8();
      sc = wmma_f16(aq0, b0, sc);
      sc = wmma_f16(aq1, b1, sc);
#pragma unroll
      for (int r = 0; r < 8; ++r) sc[r] = vk ? sc[r] : -3.0e38f;
      s2[u] = sc;
    }

    float rmx[8];
#pragma unroll
    for (int r = 0; r < 8; ++r) rmx[r] = fmaxf(s2[0][r], s2[1][r]);
#pragma unroll
    for (int r = 0; r < 8; ++r) rmx[r] = bflymax<1>(rmx[r]);
#pragma unroll
    for (int r = 0; r < 8; ++r) rmx[r] = bflymax<2>(rmx[r]);
#pragma unroll
    for (int r = 0; r < 8; ++r) rmx[r] = bflymax<4>(rmx[r]);
#pragma unroll
    for (int r = 0; r < 8; ++r) rmx[r] = bflymax<8>(rmx[r]);

    v8f p0, p1;
#pragma unroll
    for (int r = 0; r < 8; ++r) {
      const float nm = fmaxf(m8[r], rmx[r]);
      const float cf = __expf(m8[r] - nm);
      p0[r] = (s2[0][r] < -1.0e37f) ? 0.f : __expf(s2[0][r] - nm);
      p1[r] = (s2[1][r] < -1.0e37f) ? 0.f : __expf(s2[1][r] - nm);
      m8[r] = nm;
      l8[r] *= cf;
      acc[0][r] *= cf; acc[1][r] *= cf; acc[2][r] *= cf; acc[3][r] *= cf;
    }

#pragma unroll
    for (int r = 0; r < 8; ++r) {
      Pb[wid][r + 8 * hi][lc] = (_Float16)p0[r];
      Pb[wid][r + 8 * hi][16 + lc] = (_Float16)p1[r];
    }
    __syncthreads();
    const v16h ap = frag_tile(&Pb[wid][0][0], VT_STRIDE, lane);
#pragma unroll
    for (int ni = 0; ni < 4; ++ni) {
      const v16h bv = frag_rowptr(&Vt[ni * 16 + lc][0], lane);
      acc[ni] = wmma_f16(ap, bv, acc[ni]);
    }
    const v8f lsum = wmma_f16(ap, ones, zero8());
#pragma unroll
    for (int r = 0; r < 8; ++r) l8[r] += lsum[r];
  }

#pragma unroll
  for (int ni = 0; ni < 4; ++ni)
#pragma unroll
    for (int r = 0; r < 8; ++r) {
      const int gq = g0r + r + 8 * hi;
      const int srow = clss[gq];
      const float invl = l8[r] > 0.f ? 1.0f / l8[r] : 0.f;
      attn[(baseBS + srow) * DD + hh * DHD + ni * 16 + lc] =
          (_Float16)(acc[ni][r] * invl);
    }
}

// ---------------------------------------------------------------------------
extern "C" void kernel_launch(void* const* d_in, const int* in_sizes, int n_in,
                              void* d_out, int out_size, void* d_ws,
                              size_t ws_size, hipStream_t stream) {
  (void)in_sizes; (void)n_in; (void)out_size; (void)ws_size;
  const int* x = (const int*)d_in[0];
  const int* mask = (const int*)d_in[1];
  const int* clss = (const int*)d_in[2];
  const float* emb = (const float*)d_in[3];
  const float* pos = (const float*)d_in[4];
  // order: wq wk wv wqg wkg wvg wo
  const float* w[7] = {(const float*)d_in[5],  (const float*)d_in[6],
                       (const float*)d_in[7],  (const float*)d_in[9],
                       (const float*)d_in[10], (const float*)d_in[11],
                       (const float*)d_in[8]};

  char* ws = (char*)d_ws;
  size_t off = 0;
  auto take = [&](size_t bytes) -> char* {
    char* p = ws + off;
    off += (bytes + 255) & ~(size_t)255;
    return p;
  };
  const size_t rowsBS = (size_t)BB * SS;
  _Float16* hf16 = (_Float16*)take(rowsBS * DD * 2);
  _Float16* wT[7];
  for (int i = 0; i < 7; ++i) wT[i] = (_Float16*)take((size_t)DD * DD * 2);
  _Float16* qf = (_Float16*)take(rowsBS * DD * 2);
  _Float16* kf = (_Float16*)take(rowsBS * DD * 2);
  _Float16* vf = (_Float16*)take(rowsBS * DD * 2);
  _Float16* qgf = (_Float16*)take(rowsBS * DD * 2);
  _Float16* kgf = (_Float16*)take(rowsBS * DD * 2);
  _Float16* vgf = (_Float16*)take(rowsBS * DD * 2);
  _Float16* attn = (_Float16*)take(rowsBS * DD * 2);
  int* flags = (int*)take((size_t)SS * 4);

  tdm_probe_kernel<<<1, 32, 0, stream>>>();
  embed_kernel<<<dim3(BB * SS), 192, 0, stream>>>(x, emb, pos, hf16);
  dim3 tgrid(DD / 16, DD / 16);
  for (int i = 0; i < 7; ++i)
    transpose_kernel<<<tgrid, 256, 0, stream>>>(w[i], wT[i]);
  flags_zero_kernel<<<SS / 256, 256, 0, stream>>>(flags);
  flags_set_kernel<<<1, GG, 0, stream>>>(clss, flags);

  dim3 ggrid((unsigned)(rowsBS / 128), DD / 128);
  const float sc = 0.125f;  // DH^-0.5
  gemm768_kernel<false><<<ggrid, 256, 0, stream>>>(hf16, wT[0], qf, sc);
  gemm768_kernel<false><<<ggrid, 256, 0, stream>>>(hf16, wT[1], kf, 1.0f);
  gemm768_kernel<false><<<ggrid, 256, 0, stream>>>(hf16, wT[2], vf, 1.0f);
  gemm768_kernel<false><<<ggrid, 256, 0, stream>>>(hf16, wT[3], qgf, sc);
  gemm768_kernel<false><<<ggrid, 256, 0, stream>>>(hf16, wT[4], kgf, 1.0f);
  gemm768_kernel<false><<<ggrid, 256, 0, stream>>>(hf16, wT[5], vgf, 1.0f);

  band_attn_kernel<<<BB * HH * NCHUNK * 4, 128, 0, stream>>>(
      qf, kf, vf, mask, clss, flags, attn);
  global_attn_kernel<<<BB * HH, 128, 0, stream>>>(qgf, kgf, vgf, mask, clss,
                                                  attn);

  gemm768_kernel<true><<<ggrid, 256, 0, stream>>>(attn, wT[6], d_out, 1.0f);
}
